// TemplateMatchingLSTM_1683627180337
// MI455X (gfx1250) — matrix operand order repaired and checked by reading, not compiled, not run on hardware
//
#include <hip/hip_runtime.h>
#include <hip/hip_bf16.h>

// ---------------------------------------------------------------------------
// LSTM forward for MI455X (gfx1250, wave32, WMMA).
// B=64, S=1024, V=32000, H=1024, O=128.
// gates = [xe_t | h] @ [W_ih | W_hh]^T + bias  via v_wmma_f32_16x16x32_bf16.
// Embedding table pre-converted to bf16 (64 MiB, L2-resident). B tiles are
// double-buffered in LDS via async global->LDS copies (ASYNCcnt pipeline):
//   wait(prev chunk) -> barrier -> issue(next chunk, other buf) -> WMMA(cur).
// ---------------------------------------------------------------------------

typedef __attribute__((ext_vector_type(16))) __bf16 v16bf;
typedef __attribute__((ext_vector_type(8)))  float  v8f;
typedef __attribute__((ext_vector_type(4)))  int    v4i;

typedef __attribute__((address_space(1))) v4i* gas_v4i_ptr;  // global
typedef __attribute__((address_space(3))) v4i* lds_v4i_ptr;  // LDS

constexpr int kB  = 64;     // batch
constexpr int kS  = 1024;   // sequence
constexpr int kV  = 32000;  // vocab
constexpr int kH  = 1024;   // hidden
constexpr int kG  = 4096;   // 4*H gate rows
constexpr int kK  = 2048;   // fused K = H(x) + H(h)
constexpr int kNC = kK / 32;// K-chunks
constexpr int kO  = 128;    // output classes

struct U4x2 { uint4 lo, hi; };   // 32 bytes == v16bf

#if __has_builtin(__builtin_amdgcn_global_load_async_to_lds_b128)
#define USE_ASYNC_LDS 1
#else
#define USE_ASYNC_LDS 0
#endif

__device__ __forceinline__ unsigned short f2bf_bits(float f) {
  unsigned u = __builtin_bit_cast(unsigned, f);
  u += 0x7FFFu + ((u >> 16) & 1u);          // round-to-nearest-even
  return (unsigned short)(u >> 16);
}

__device__ __forceinline__ float sigmoidf_(float x) {
  return 1.0f / (1.0f + __expf(-x));
}

__device__ __forceinline__ void wait_async_zero() {
#if __has_builtin(__builtin_amdgcn_s_wait_asynccnt)
  __builtin_amdgcn_s_wait_asynccnt(0);
#else
  asm volatile("s_wait_asynccnt 0x0" ::: "memory");
#endif
}

#if USE_ASYNC_LDS
// 16-byte async copy global -> LDS (tracked by ASYNCcnt).
__device__ __forceinline__ void async_copy_b128(const void* g, void* l) {
  __builtin_amdgcn_global_load_async_to_lds_b128(
      (gas_v4i_ptr)(v4i*)const_cast<void*>(g), (lds_v4i_ptr)(v4i*)l, 0, 0);
}
#endif

// ---------------------------------------------------------------------------
// Prep kernels
// ---------------------------------------------------------------------------
__global__ void prep_weights_kernel(const float* __restrict__ W_ih,
                                    const float* __restrict__ W_hh,
                                    unsigned short* __restrict__ Wc) {
  const int total = kG * kK;
  for (int e = blockIdx.x * blockDim.x + threadIdx.x; e < total;
       e += gridDim.x * blockDim.x) {
    const int g = e >> 11;          // / kK
    const int k = e & (kK - 1);
    const float v = (k < kH) ? W_ih[g * kH + k] : W_hh[g * kH + (k - kH)];
    Wc[e] = f2bf_bits(v);
  }
}

__global__ void prep_emb_kernel(const float* __restrict__ emb,
                                unsigned short* __restrict__ embbf) {
  const int total = kV * kH;
  for (int e = blockIdx.x * blockDim.x + threadIdx.x; e < total;
       e += gridDim.x * blockDim.x)
    embbf[e] = f2bf_bits(emb[e]);
}

__global__ void prep_state_kernel(const float* __restrict__ b_ih,
                                  const float* __restrict__ b_hh,
                                  float* __restrict__ bias,
                                  const float* __restrict__ h0,
                                  const float* __restrict__ c0,
                                  float* __restrict__ h_f32,
                                  float* __restrict__ c_f32,
                                  unsigned short* __restrict__ hbf0) {
  const int e = blockIdx.x * blockDim.x + threadIdx.x;
  if (e < kG) bias[e] = b_ih[e] + b_hh[e];
  if (e < kB * kH) {
    const float h = h0[e];
    h_f32[e] = h;
    c_f32[e] = c0[e];
    hbf0[e]  = f2bf_bits(h);
  }
}

// ---------------------------------------------------------------------------
// One LSTM timestep.
// Grid: 64 blocks (N-tiles of 16 cols). Block: 128 threads = 4 waves
// (M-tiles of 16 batch rows). Each wave accumulates 4 gate tiles (i,f,g,o)
// over K=2048 in chunks of 32, 4 WMMAs per chunk sharing one A fragment.
// ---------------------------------------------------------------------------
__global__ __launch_bounds__(128)
void lstm_step_kernel(const int*            __restrict__ x,
                      const unsigned short* __restrict__ embbf,
                      const unsigned short* __restrict__ Wc,
                      const float*          __restrict__ bias,
                      const unsigned short* __restrict__ hprev,
                      unsigned short*       __restrict__ hnext,
                      float*                __restrict__ c_f32,
                      float*                __restrict__ h_f32,
                      int t) {
  // 2 x 4 KiB ping-pong: bf16 [gate][n:16][k:32], k fastest
  __shared__ uint4 ldsB4[2][256];

  const int tid   = threadIdx.x;
  const int lane  = tid & 31;
  const int wave  = tid >> 5;       // M-tile (batch rows wave*16 .. +15)
  const int nt    = blockIdx.x;     // N-tile
  const int row   = lane & 15;
  const int khalf = lane >> 4;      // 0 or 1
  const int ntbase = nt * 16;

  // A-matrix row for this lane (batch b = wave*16 + row; 16-bit A layout:
  // lanes 0-15 hold K {0..7,16..23}, lanes 16-31 hold K {8..15,24..31}).
  const int b = wave * 16 + row;
  const int embIdx = x[b * kS + t];
  const unsigned short* erow = embbf + (size_t)embIdx * kH;
  const unsigned short* hrow = hprev + (size_t)b * kH;

  // Stage one 4-KiB B chunk (4 gates x 32x16 bf16) cooperatively.
  auto stage = [&](int kc) {
    const int k0   = kc * 32;
    const int bufi = kc & 1;
    #pragma unroll
    for (int rep = 0; rep < 2; ++rep) {
      const int s = tid + rep * 128;            // 0..255 uint4 slots
      const int g    = s >> 6;
      const int n    = (s >> 2) & 15;
      const int ksub = (s & 3) * 8;
      const size_t goff = (size_t)(g * kH + ntbase + n) * kK + k0 + ksub;
#if USE_ASYNC_LDS
      async_copy_b128(Wc + goff, &ldsB4[bufi][s]);
#else
      ldsB4[bufi][s] = *(const uint4*)(Wc + goff);
#endif
      // warm GL2 one chunk further ahead (global_prefetch_b8)
      if (kc + 1 < kNC) __builtin_prefetch(Wc + goff + 32, 0, 1);
    }
  };

  v8f acc[4];
  #pragma unroll
  for (int g = 0; g < 4; ++g) acc[g] = v8f{};

  stage(0);   // prologue

  #pragma unroll 1
  for (int kc = 0; kc < kNC; ++kc) {
    // My async writes for chunk kc have landed; barrier makes everyone's
    // visible and guarantees buffer (kc+1)&1 is no longer being read.
#if USE_ASYNC_LDS
    wait_async_zero();
#endif
    __syncthreads();
    if (kc + 1 < kNC) stage(kc + 1);   // overlaps with compute below

    const int k0 = kc * 32;
    const uint4* buf = ldsB4[kc & 1];

    // ---- A fragment (16x32 bf16), all-bf16 sources ----
    const int base1 = k0 + khalf * 8;
    const int base2 = k0 + 16 + khalf * 8;
    const unsigned short* ar = (k0 < kH) ? erow : (hrow - kH);
    U4x2 u;
    u.lo = *(const uint4*)(ar + base1);
    u.hi = *(const uint4*)(ar + base2);
    const v16bf a = __builtin_bit_cast(v16bf, u);

    // ---- 4 WMMAs (one per gate), B from LDS per K-striped B layout ----
    #pragma unroll
    for (int g = 0; g < 4; ++g) {
      const int bi = (g * 16 + row) * 4 + khalf;   // uint4 index
      U4x2 ub;
      ub.lo = buf[bi];
      ub.hi = buf[bi + 2];
      const v16bf bv = __builtin_bit_cast(v16bf, ub);
      acc[g] = __builtin_amdgcn_wmma_f32_16x16x32_bf16(
          false, a, false, bv, (short)0, acc[g], false, false);
    }
  }

  // ---- epilogue: activations + cell update.
  // C/D layout: VGPR r -> lanes 0-15: M=r, lanes 16-31: M=8+r; N = lane&15.
  const int n = ntbase + row;
  const float bi_i = bias[0 * kH + n];
  const float bi_f = bias[1 * kH + n];
  const float bi_g = bias[2 * kH + n];
  const float bi_o = bias[3 * kH + n];

  #pragma unroll
  for (int r = 0; r < 8; ++r) {
    const int    bb  = wave * 16 + khalf * 8 + r;
    const size_t off = (size_t)bb * kH + n;
    const float iv = sigmoidf_(acc[0][r] + bi_i);
    const float fv = sigmoidf_(acc[1][r] + bi_f);
    const float gv = tanhf(acc[2][r] + bi_g);
    const float ov = sigmoidf_(acc[3][r] + bi_o);
    const float cn = fv * c_f32[off] + iv * gv;
    c_f32[off] = cn;
    const float hn = ov * tanhf(cn);
    h_f32[off] = hn;
    hnext[off] = f2bf_bits(hn);
  }
}

// ---------------------------------------------------------------------------
// Final FC: out[b][o] = h_last[b] . fc_W[o] + fc_b[o]   (64x128x1024, tiny)
// ---------------------------------------------------------------------------
__global__ __launch_bounds__(128)
void fc_kernel(const float* __restrict__ h,
               const float* __restrict__ fc_W,
               const float* __restrict__ fc_b,
               float* __restrict__ out) {
  const int o = threadIdx.x;   // 0..127
  const int b = blockIdx.x;    // 0..63
  const float* hr = h    + (size_t)b * kH;
  const float* wr = fc_W + (size_t)o * kH;
  float s = 0.0f;
  for (int k = 0; k < kH; k += 4) {
    const float4 hv = *(const float4*)(hr + k);
    const float4 wv = *(const float4*)(wr + k);
    s += hv.x * wv.x + hv.y * wv.y + hv.z * wv.z + hv.w * wv.w;
  }
  out[b * kO + o] = s + fc_b[o];
}

// ---------------------------------------------------------------------------
extern "C" void kernel_launch(void* const* d_in, const int* in_sizes, int n_in,
                              void* d_out, int out_size, void* d_ws, size_t ws_size,
                              hipStream_t stream) {
  const int*   x    = (const int*)  d_in[0];
  const float* emb  = (const float*)d_in[1];
  const float* W_ih = (const float*)d_in[2];
  const float* W_hh = (const float*)d_in[3];
  const float* b_ih = (const float*)d_in[4];
  const float* b_hh = (const float*)d_in[5];
  const float* fc_W = (const float*)d_in[6];
  const float* fc_b = (const float*)d_in[7];
  const float* h0   = (const float*)d_in[8];
  const float* c0   = (const float*)d_in[9];
  float* out = (float*)d_out;

  // Workspace carve-up (~82 MB total, all 16B-aligned offsets)
  char* ws = (char*)d_ws;
  unsigned short* embbf = (unsigned short*)ws; ws += (size_t)kV * kH * 2;  // 64 MiB
  unsigned short* Wc    = (unsigned short*)ws; ws += (size_t)kG * kK * 2;  // 16 MiB
  float*          bias  = (float*)ws;          ws += (size_t)kG * 4;
  unsigned short* hbf0  = (unsigned short*)ws; ws += (size_t)kB * kH * 2;
  unsigned short* hbf1  = (unsigned short*)ws; ws += (size_t)kB * kH * 2;
  float*          cbuf  = (float*)ws;          ws += (size_t)kB * kH * 4;
  float*          hbuf  = (float*)ws;          ws += (size_t)kB * kH * 4;

  prep_weights_kernel<<<512, 256, 0, stream>>>(W_ih, W_hh, Wc);
  prep_emb_kernel<<<2048, 256, 0, stream>>>(emb, embbf);
  prep_state_kernel<<<(kB * kH + 255) / 256, 256, 0, stream>>>(
      b_ih, b_hh, bias, h0, c0, hbuf, cbuf, hbf0);

  unsigned short* hb[2] = {hbf0, hbf1};
  for (int t = 0; t < kS; ++t) {
    lstm_step_kernel<<<kH / 16, 128, 0, stream>>>(
        x, embbf, Wc, bias, hb[t & 1], hb[(t + 1) & 1], cbuf, hbuf, t);
  }

  fc_kernel<<<kB, kO, 0, stream>>>(hbuf, fc_W, fc_b, out);
}
